// CGModel_29506425324140
// MI455X (gfx1250) — matrix operand order repaired and checked by reading, not compiled, not run on hardware
//
#include <hip/hip_runtime.h>
#include <hip/hip_bf16.h>
#include <math.h>

// ---------------- problem constants ----------------
#define NSV 48
#define NVV 10
#define BQ  4
#define NLV 160
#define NRV 1200
#define NNV (NLV + NRV)            // 1360
#define ELV 2560
#define ERV 9600
#define ECV 9600
#define ETV (ELV + ERV + 2*ECV)    // 31360 = 490*64
#define XD  108                    // NS + 6*NV
#define KP  160                    // padded GEMM K (144 -> 160, mult of 32)
#define NP1 160                    // padded hidden (144 -> 160)
#define NW0 2784                   // conv0 weight width (mult of 32)
#define NW1V 3464
#define NW1P 3488                  // padded to mult of 32
#define CHUNKE 2240                // edges per w-chunk (mult of 64); 31360/2240=14
#define NCHUNK 14

#define INV_SQRT48 0.14433756729740645f
#define INV_SQRT10 0.31622776601683794f
#define INV_SQRT2  0.7071067811865476f
#define INV_SQRT3  0.5773502691896258f
#define SQRT3F     1.7320508075688772f

// ---------------- bf16 helpers ----------------
typedef __attribute__((ext_vector_type(16))) __bf16 v16bf;
typedef __attribute__((ext_vector_type(8)))  float  v8f;

__device__ __forceinline__ unsigned short f32_to_bf16_bits(float f) {
    unsigned int u = __float_as_uint(f);
    unsigned int lsb = (u >> 16) & 1u;
    u += 0x7fffu + lsb;              // round-to-nearest-even
    return (unsigned short)(u >> 16);
}

__device__ __forceinline__ float rbf_val(float dist, float dmax, int j) {
    float step  = dmax / 31.0f;                 // linspace(0,dmax,32)
    float coeff = -0.5f / (step * step);
    float dd = dist - step * (float)j;
    return __expf(coeff * dd * dd);
}

// ---- CDNA5 async global->LDS DMA (ASYNCcnt-tracked), inline asm (portable path) ----
__device__ __forceinline__ void async_b128_to_lds(const void* gsrc, void* lds_dst) {
    // generic pointer to LDS: low 32 bits are the LDS byte address (flat aperture rule)
    unsigned off = (unsigned)(unsigned long long)lds_dst;
    asm volatile("global_load_async_to_lds_b128 %0, %1, off"
                 :: "v"(off), "v"(gsrc) : "memory");
}
__device__ __forceinline__ void wait_async0() {
    asm volatile("s_wait_asynccnt 0x0" ::: "memory");
}

// ---------------- generic small kernels ----------------
__global__ void k_zero(float* p, int n) {
    int i = blockIdx.x * blockDim.x + threadIdx.x;
    if (i < n) p[i] = 0.0f;
}
__global__ void k_copy(float* dst, const float* src, int n) {
    int i = blockIdx.x * blockDim.x + threadIdx.x;
    if (i < n) dst[i] = src[i];
}
// out[m,n] = act(bias[n] + sum_k in[m,k]*W[k,n])
__global__ void k_linear(const float* __restrict__ in, const float* __restrict__ W,
                         const float* __restrict__ b, float* __restrict__ out,
                         int M, int K, int N, int relu) {
    int idx = blockIdx.x * blockDim.x + threadIdx.x;
    if (idx >= M * N) return;
    int m = idx / N, n = idx % N;
    float a = b[n];
    const float* ir = in + (size_t)m * K;
    for (int k = 0; k < K; ++k) a += ir[k] * W[(size_t)k * N + n];
    if (relu) a = fmaxf(a, 0.0f);
    out[idx] = a;
}

// ---------------- per-batch sigma vectors ----------------
__global__ void k_batch_vecs(const float* __restrict__ sig,
                             const float* __restrict__ lw, const float* __restrict__ lb,
                             const float* __restrict__ rw1, const float* __restrict__ rb1,
                             const float* __restrict__ rw2, const float* __restrict__ rb2,
                             float* __restrict__ ligv, float* __restrict__ recv) {
    __shared__ float h[BQ * 48];
    int t = threadIdx.x;
    if (t < BQ * 48) {
        int bb = t / 48, j = t % 48;
        float a = lb[j], r = rb1[j];
        for (int k = 0; k < 32; ++k) {
            float s = sig[bb * 32 + k];
            a += s * lw[k * 48 + j];
            r += s * rw1[k * 48 + j];
        }
        ligv[t] = a;
        h[t] = fmaxf(r, 0.0f);
    }
    __syncthreads();
    if (t < BQ * 48) {
        int bb = t / 48, j = t % 48;
        float a = rb2[j];
        for (int k = 0; k < 48; ++k) a += h[bb * 48 + k] * rw2[k * 48 + j];
        recv[t] = a;
    }
}

// ---------------- node init ----------------
__global__ void k_node_init(const float* __restrict__ lig_emb, const float* __restrict__ rec_emb,
                            const int* __restrict__ lig_idx, const int* __restrict__ rec_idx,
                            const int* __restrict__ lig_batch, const int* __restrict__ rec_batch,
                            const float* __restrict__ ligv, const float* __restrict__ recv,
                            float* __restrict__ x) {
    int idx = blockIdx.x * blockDim.x + threadIdx.x;
    if (idx >= NNV * XD) return;
    int n = idx / XD, c = idx % XD;
    float v = 0.0f;
    if (c < NSV) {
        if (n < NLV) v = lig_emb[lig_idx[n] * NSV + c] + ligv[lig_batch[n] * NSV + c];
        else {
            int m = n - NLV;
            v = rec_emb[rec_idx[m] * NSV + c] + recv[rec_batch[m] * NSV + c];
        }
    }
    x[idx] = v;
}

// ---------------- edge geometry + counts ----------------
__global__ void k_edges(const float* __restrict__ lig_pos, const float* __restrict__ rec_pos,
                        const int* __restrict__ lig_ei, const int* __restrict__ rec_ei,
                        const int* __restrict__ cross_ei,
                        int* __restrict__ src, int* __restrict__ dst,
                        float* __restrict__ darr, float* __restrict__ sh1,
                        float* __restrict__ cnt) {
    int e = blockIdx.x * blockDim.x + threadIdx.x;
    if (e >= ETV) return;
    int s, d;
    if (e < ELV)                   { s = lig_ei[e];                 d = lig_ei[ELV + e]; }
    else if (e < ELV + ERV)        { int t = e - ELV;               s = rec_ei[t] + NLV;       d = rec_ei[ERV + t] + NLV; }
    else if (e < ELV + ERV + ECV)  { int t = e - ELV - ERV;         s = cross_ei[t] + NLV;     d = cross_ei[ECV + t]; }
    else                           { int t = e - ELV - ERV - ECV;   s = cross_ei[ECV + t];     d = cross_ei[t] + NLV; }
    src[e] = s; dst[e] = d;
    const float* ps = (s < NLV) ? (lig_pos + 3 * s) : (rec_pos + 3 * (s - NLV));
    const float* pd = (d < NLV) ? (lig_pos + 3 * d) : (rec_pos + 3 * (d - NLV));
    float vx = pd[0] - ps[0], vy = pd[1] - ps[1], vz = pd[2] - ps[2];
    float r = sqrtf(vx * vx + vy * vy + vz * vz);
    darr[e] = r;
    float inv = SQRT3F / (r + 1e-8f);
    sh1[3 * e] = vx * inv; sh1[3 * e + 1] = vy * inv; sh1[3 * e + 2] = vz * inv;
    atomicAdd(&cnt[d], 1.0f);
}

// ---------------- edge MLP input builders ----------------
__global__ void k_in_l(const float* __restrict__ eattr, const float* __restrict__ sig,
                       const int* __restrict__ lig_ei, const int* __restrict__ lig_batch,
                       const float* __restrict__ darr, float* __restrict__ out) {
    int e = blockIdx.x * blockDim.x + threadIdx.x;
    if (e >= ELV) return;
    int b = lig_batch[lig_ei[ELV + e]];
    float dd = darr[e];
    float* o = out + (size_t)e * 68;
    for (int j = 0; j < 4;  ++j) o[j]      = eattr[e * 4 + j];
    for (int j = 0; j < 32; ++j) o[4 + j]  = sig[b * 32 + j];
    for (int j = 0; j < 32; ++j) o[36 + j] = rbf_val(dd, 5.0f, j);
}
__global__ void k_in_r(const float* __restrict__ darr, float* __restrict__ out) {
    int e = blockIdx.x * blockDim.x + threadIdx.x;
    if (e >= ERV) return;
    float dd = darr[ELV + e];
    float* o = out + (size_t)e * 32;
    for (int j = 0; j < 32; ++j) o[j] = rbf_val(dd, 30.0f, j);
}
__global__ void k_in_c(const float* __restrict__ sig, const int* __restrict__ cross_ei,
                       const int* __restrict__ lig_batch, const float* __restrict__ darr,
                       float* __restrict__ out) {
    int e = blockIdx.x * blockDim.x + threadIdx.x;
    if (e >= ECV) return;
    int b = lig_batch[cross_ei[ECV + e]];
    float dd = darr[ELV + ERV + e];
    float* o = out + (size_t)e * 64;
    for (int j = 0; j < 32; ++j) o[j]      = sig[b * 32 + j];
    for (int j = 0; j < 32; ++j) o[32 + j] = rbf_val(dd, 250.0f, j);
}

// ---------------- bf16 staging ----------------
// ef[e, 0:48]=ea, [48:96]=x[src][:48], [96:144]=x[dst][:48], [144:160]=0
// 8 bf16 per thread, one b128 store (48/96/144 are multiples of 8)
__global__ void k_stage_ef(const float* __restrict__ ea, const float* __restrict__ x,
                           const int* __restrict__ src, const int* __restrict__ dst,
                           unsigned short* __restrict__ ef16) {
    long idx = (long)blockIdx.x * blockDim.x + threadIdx.x;
    if (idx >= (long)ETV * (KP / 8)) return;
    int e  = (int)(idx / (KP / 8));
    int k8 = (int)(idx % (KP / 8)) * 8;
    const float* sp = nullptr;
    int base = 0;
    if (k8 < 48)       { sp = ea + (size_t)e * NSV;      base = k8; }
    else if (k8 < 96)  { sp = x + (size_t)src[e] * XD;   base = k8 - 48; }
    else if (k8 < 144) { sp = x + (size_t)dst[e] * XD;   base = k8 - 96; }
    unsigned int u[4];
    for (int j = 0; j < 4; ++j) {
        unsigned short lo = sp ? f32_to_bf16_bits(sp[base + 2 * j])     : (unsigned short)0;
        unsigned short hi = sp ? f32_to_bf16_bits(sp[base + 2 * j + 1]) : (unsigned short)0;
        u[j] = (unsigned int)lo | ((unsigned int)hi << 16);
    }
    uint4 pk; pk.x = u[0]; pk.y = u[1]; pk.z = u[2]; pk.w = u[3];
    *reinterpret_cast<uint4*>(ef16 + (size_t)e * KP + k8) = pk;
}
// stage W [K,N] f32 -> bf16 FRAGMENT-MAJOR: flat = ((kt*ntiles + ntile)*32 + lane)*16 + t
// with k = kt*32 + (lane>>4)*8 + (t&7) + (t>>3)*16 ; n = ntile*16 + (lane&15)
// -> in the GEMM, each lane's 16 B-fragment elements are 32 contiguous bytes.
__global__ void k_stage_w(const float* __restrict__ W, unsigned short* __restrict__ W16,
                          int K, int N, int Npad) {
    int idx = blockIdx.x * blockDim.x + threadIdx.x;
    if (idx >= KP * Npad) return;
    int t      = idx & 15;
    int lane   = (idx >> 4) & 31;
    int blk    = idx >> 9;                 // kt*ntiles + ntile
    int ntiles = Npad >> 4;
    int kt = blk / ntiles, ntile = blk % ntiles;
    int k = kt * 32 + (lane >> 4) * 8 + (t & 7) + (t >> 3) * 16;
    int n = ntile * 16 + (lane & 15);
    float v = (k < K && n < N) ? W[(size_t)k * N + n] : 0.0f;
    W16[idx] = f32_to_bf16_bits(v);
}
__global__ void k_pad_bias(const float* __restrict__ b, float* __restrict__ bp, int N, int Npad) {
    int i = blockIdx.x * blockDim.x + threadIdx.x;
    if (i < Npad) bp[i] = (i < N) ? b[i] : 0.0f;
}

// ---------------- WMMA bf16 GEMM: C = act(A@B + bias) ----------------
// A bf16 [M, lda] row-major, staged to LDS via GLOBAL_LOAD_ASYNC_TO_LDS_B128
// (ASYNCcnt-tracked DMA, double buffered, overlapped with WMMA);
// Bs bf16 fragment-major (see k_stage_w) -> two coalesced b128 loads per lane, no LDS.
// Block = 64M x 32N, 8 waves as 4(M) x 2(N) 16x16 tiles.
__global__ __launch_bounds__(256) void k_wmma_gemm(
    const unsigned short* __restrict__ A, int lda,
    const unsigned short* __restrict__ Bs, int ntiles,
    const float* __restrict__ bias,
    float* __restrict__ Cf, unsigned short* __restrict__ Ch, int ldc,
    int relu, int out_bf16)
{
    __shared__ __align__(16) unsigned short At[2][64 * 32];
    int tid  = threadIdx.x;
    int lane = tid & 31;
    int wid  = tid >> 5;
    int mw   = wid & 3;
    int nw   = wid >> 2;
    long mBlock = (long)blockIdx.y * 64;
    int  ntile  = blockIdx.x * 2 + nw;
    int r = tid >> 2, q = tid & 3;          // A-tile staging role: row r, 8-elem chunk q

    union ABu { struct { uint4 lo, hi; } qv; v16bf v; };

    // prologue: async-DMA k0 = 0 tile straight into LDS
    async_b128_to_lds(A + (mBlock + r) * (long)lda + q * 8, &At[0][r * 32 + q * 8]);
    wait_async0();
    __syncthreads();

    v8f acc = {};
    int mrow  = mw * 16 + (lane & 15);
    int khalf = lane >> 4;
#pragma unroll
    for (int i = 0; i < KP / 32; ++i) {
        // kick off DMA of the next A tile into the other buffer (overlaps with WMMA below)
        if (i + 1 < KP / 32) {
            async_b128_to_lds(A + (mBlock + r) * (long)lda + (i + 1) * 32 + q * 8,
                              &At[(i + 1) & 1][r * 32 + q * 8]);
        }
        // B fragment: 32 contiguous bytes per lane from fragment-major buffer
        ABu bu;
        const uint4* bp = reinterpret_cast<const uint4*>(
            Bs + (((size_t)i * ntiles + ntile) << 9) + (size_t)lane * 16);
        bu.qv.lo = bp[0]; bu.qv.hi = bp[1];
        // A fragment: lane<16 K={0..7,16..23}, lane>=16 +8 -> two aligned b128 LDS loads
        ABu au;
        const unsigned short* ab = &At[i & 1][mrow * 32 + khalf * 8];
        au.qv.lo = *reinterpret_cast<const uint4*>(ab);
        au.qv.hi = *reinterpret_cast<const uint4*>(ab + 16);
        acc = __builtin_amdgcn_wmma_f32_16x16x32_bf16(false, au.v, false, bu.v,
                                                      (short)0, acc, false, false);
        if (i + 1 < KP / 32) wait_async0();   // own DMA complete; barrier covers the rest
        __syncthreads();
    }
    int  ncol  = ntile * 16 + (lane & 15);
    float bv   = bias[ncol];
    long mBase = mBlock + mw * 16 + (khalf << 3);   // C layout: lane half splits M at 8
#pragma unroll
    for (int rr = 0; rr < 8; ++rr) {
        float v = acc[rr] + bv;
        if (relu) v = fmaxf(v, 0.0f);
        long off = (mBase + rr) * (long)ldc + ncol;
        if (out_bf16) Ch[off] = f32_to_bf16_bits(v);
        else          Cf[off] = v;
    }
}

// ---------------- tensor products (scatter-accumulate into agg) ----------------
__global__ void k_tp0(const float* __restrict__ w, int wstride, int ebase,
                      const float* __restrict__ x, const int* __restrict__ src,
                      const int* __restrict__ dst, const float* __restrict__ sh1,
                      float* __restrict__ agg) {
    int e = ebase + blockIdx.x;
    const float* wr = w + (size_t)blockIdx.x * wstride;
    __shared__ float x0[48];
    __shared__ float s1[3];
    int t = threadIdx.x;               // 96 threads
    int s = src[e], d = dst[e];
    if (t < 48) x0[t] = x[(size_t)s * XD + t];
    if (t >= 48 && t < 51) s1[t - 48] = sh1[3 * e + (t - 48)];
    __syncthreads();
    if (t < 48) {
        float a = 0.0f;
        for (int i = 0; i < 48; ++i) a += x0[i] * wr[i * 48 + t];
        atomicAdd(&agg[(size_t)d * XD + t], a * INV_SQRT48);
    } else if (t < 78) {
        int j = (t - 48) / 3, k = (t - 48) % 3;
        float a = 0.0f;
        for (int i = 0; i < 48; ++i) a += x0[i] * wr[2304 + i * 10 + j];
        atomicAdd(&agg[(size_t)d * XD + t], a * s1[k] * INV_SQRT48);
    }
}

__global__ void k_tp1(const float* __restrict__ w, int wstride, int ebase,
                      const float* __restrict__ x, const int* __restrict__ src,
                      const int* __restrict__ dst, const float* __restrict__ sh1,
                      float* __restrict__ agg) {
    int e = ebase + blockIdx.x;
    const float* wr = w + (size_t)blockIdx.x * wstride;
    __shared__ float x0[48];
    __shared__ float xv[30];
    __shared__ float s1[3];
    __shared__ float dp[10];
    __shared__ float cr[30];
    int t = threadIdx.x;               // 128 threads
    int s = src[e], d = dst[e];
    if (t < 48) x0[t] = x[(size_t)s * XD + t];
    else if (t < 78) xv[t - 48] = x[(size_t)s * XD + t];
    else if (t < 81) s1[t - 78] = sh1[3 * e + (t - 78)];
    __syncthreads();
    if (t < 10) {
        dp[t] = (xv[t * 3] * s1[0] + xv[t * 3 + 1] * s1[1] + xv[t * 3 + 2] * s1[2]) * INV_SQRT3;
    } else if (t < 20) {
        int i = t - 10;
        float ax = xv[i * 3], ay = xv[i * 3 + 1], az = xv[i * 3 + 2];
        cr[i * 3 + 0] = (ay * s1[2] - az * s1[1]) * INV_SQRT2;
        cr[i * 3 + 1] = (az * s1[0] - ax * s1[2]) * INV_SQRT2;
        cr[i * 3 + 2] = (ax * s1[1] - ay * s1[0]) * INV_SQRT2;
    }
    __syncthreads();
    if (t < 48) {
        float a = 0.0f, b = 0.0f;
        for (int i = 0; i < 48; ++i) a += x0[i] * wr[i * 48 + t];
        for (int i = 0; i < 10; ++i) b += dp[i] * wr[2304 + i * 48 + t];
        float v = (a * INV_SQRT48 + b * INV_SQRT10) * INV_SQRT2;
        atomicAdd(&agg[(size_t)d * XD + t], v);
    } else if (t < 78) {
        int j = (t - 48) / 3, k = (t - 48) % 3;
        float c = 0.0f, dv = 0.0f;
        for (int i = 0; i < 48; ++i) c  += x0[i] * wr[2784 + i * 10 + j];
        for (int i = 0; i < 10; ++i) dv += xv[i * 3 + k] * wr[3264 + i * 10 + j];
        float v = (c * s1[k] * INV_SQRT48 + dv * INV_SQRT10) * INV_SQRT2;
        atomicAdd(&agg[(size_t)d * XD + t], v);
    } else if (t < 108) {
        int j = (t - 78) / 3, k = (t - 78) % 3;
        float v = 0.0f;
        for (int i = 0; i < 10; ++i) v += cr[i * 3 + k] * wr[3364 + i * 10 + j];
        atomicAdd(&agg[(size_t)d * XD + t], v * INV_SQRT10);
    }
}

__global__ void k_xupdate(float* __restrict__ x, const float* __restrict__ agg,
                          const float* __restrict__ cnt) {
    int idx = blockIdx.x * blockDim.x + threadIdx.x;
    if (idx >= NNV * XD) return;
    x[idx] += agg[idx] / fmaxf(cnt[idx / XD], 1.0f);
}

// ---------------- final (ligand-only) stage ----------------
__global__ void k_center_acc(const float* __restrict__ lig_pos, const int* __restrict__ lig_batch,
                             float* __restrict__ center, float* __restrict__ ccnt) {
    int i = blockIdx.x * blockDim.x + threadIdx.x;
    if (i >= NLV) return;
    int b = lig_batch[i];
    atomicAdd(&center[b * 3 + 0], lig_pos[i * 3 + 0]);
    atomicAdd(&center[b * 3 + 1], lig_pos[i * 3 + 1]);
    atomicAdd(&center[b * 3 + 2], lig_pos[i * 3 + 2]);
    atomicAdd(&ccnt[b], 1.0f);
}
__global__ void k_center_div(float* __restrict__ center, const float* __restrict__ ccnt) {
    int i = threadIdx.x;
    if (i < BQ * 3) center[i] /= fmaxf(ccnt[i / 3], 1.0f);
}
__global__ void k_final_feats(const float* __restrict__ lig_pos, const int* __restrict__ lig_batch,
                              const float* __restrict__ center, const float* __restrict__ sig,
                              float* __restrict__ cs1, float* __restrict__ fin_in) {
    int i = blockIdx.x * blockDim.x + threadIdx.x;
    if (i >= NLV) return;
    int b = lig_batch[i];
    float vx = center[b * 3 + 0] - lig_pos[i * 3 + 0];
    float vy = center[b * 3 + 1] - lig_pos[i * 3 + 1];
    float vz = center[b * 3 + 2] - lig_pos[i * 3 + 2];
    float r = sqrtf(vx * vx + vy * vy + vz * vz);
    float inv = SQRT3F / (r + 1e-8f);
    cs1[i * 3 + 0] = vx * inv; cs1[i * 3 + 1] = vy * inv; cs1[i * 3 + 2] = vz * inv;
    float* o = fin_in + (size_t)i * 64;
    for (int j = 0; j < 32; ++j) o[j]      = rbf_val(r, 30.0f, j);
    for (int j = 0; j < 32; ++j) o[32 + j] = sig[b * 32 + j];
}
__global__ void k_wf_in(const float* __restrict__ cea, const float* __restrict__ x,
                        float* __restrict__ wf_in) {
    int idx = blockIdx.x * blockDim.x + threadIdx.x;
    if (idx >= NLV * 96) return;
    int i = idx / 96, c = idx % 96;
    wf_in[idx] = (c < 48) ? cea[(size_t)i * 48 + c] : x[(size_t)i * XD + (c - 48)];
}
__global__ void k_tpf(const float* __restrict__ x, const float* __restrict__ cs1,
                      const int* __restrict__ lig_batch, const float* __restrict__ wf,
                      float* __restrict__ gacc, float* __restrict__ gcnt) {
    int n = blockIdx.x;
    int t = threadIdx.x;                // 12 threads
    const float* xr = x + (size_t)n * XD;
    const float* w  = wf + (size_t)n * 176;
    float s0 = cs1[n * 3], s1v = cs1[n * 3 + 1], s2 = cs1[n * 3 + 2];
    int b = lig_batch[n];
    float val;
    if (t < 6) {
        int j = t / 3, k = t % 3;
        float sk = (k == 0 ? s0 : (k == 1 ? s1v : s2));
        float f = 0.0f;
        for (int i = 0; i < 48; ++i) f += xr[i] * w[i * 2 + j];
        f *= sk * INV_SQRT48;
        float g = 0.0f, h = 0.0f;
        for (int i = 0; i < 10; ++i) {
            g += xr[48 + i * 3 + k] * w[96 + i * 2 + j];
            float ax = xr[78 + i * 3], ay = xr[78 + i * 3 + 1], az = xr[78 + i * 3 + 2];
            float cx = ay * s2 - az * s1v, cy = az * s0 - ax * s2, cz = ax * s1v - ay * s0;
            float ck = (k == 0 ? cx : (k == 1 ? cy : cz)) * INV_SQRT2;
            h += ck * w[116 + i * 2 + j];
        }
        val = (f + g * INV_SQRT10 + h * INV_SQRT10) * INV_SQRT3;
    } else {
        int j = (t - 6) / 3, k = (t - 6) % 3;
        float ii = 0.0f, jj = 0.0f;
        for (int i = 0; i < 10; ++i) {
            float ax = xr[48 + i * 3], ay = xr[48 + i * 3 + 1], az = xr[48 + i * 3 + 2];
            float cx = ay * s2 - az * s1v, cy = az * s0 - ax * s2, cz = ax * s1v - ay * s0;
            float ck = (k == 0 ? cx : (k == 1 ? cy : cz)) * INV_SQRT2;
            ii += ck * w[136 + i * 2 + j];
            jj += xr[78 + i * 3 + k] * w[156 + i * 2 + j];
        }
        val = (ii + jj) * INV_SQRT10 * INV_SQRT2;
    }
    atomicAdd(&gacc[b * 12 + t], val);
    if (t == 0) atomicAdd(&gcnt[b], 1.0f);
}
__global__ void k_heads(const float* __restrict__ gacc, const float* __restrict__ gcnt,
                        const float* __restrict__ sig,
                        const float* __restrict__ twl1, const float* __restrict__ tbl1,
                        const float* __restrict__ twl2, const float* __restrict__ tbl2,
                        const float* __restrict__ rwl1, const float* __restrict__ rbl1,
                        const float* __restrict__ rwl2, const float* __restrict__ rbl2,
                        float* __restrict__ out) {
    __shared__ float g[BQ * 12];
    __shared__ float tr[BQ * 3], rot[BQ * 3], trn[BQ], rotn[BQ];
    __shared__ float hid[2 * BQ * 48];
    __shared__ float scale[2 * BQ];
    int t = threadIdx.x;
    if (t < BQ * 12) g[t] = gacc[t] / fmaxf(gcnt[t / 12], 1.0f);
    __syncthreads();
    if (t < BQ) {
        float tx = g[t*12+0]+g[t*12+6], ty = g[t*12+1]+g[t*12+7], tz = g[t*12+2]+g[t*12+8];
        float rx = g[t*12+3]+g[t*12+9], ry = g[t*12+4]+g[t*12+10], rz = g[t*12+5]+g[t*12+11];
        tr[t*3]=tx; tr[t*3+1]=ty; tr[t*3+2]=tz;
        rot[t*3]=rx; rot[t*3+1]=ry; rot[t*3+2]=rz;
        trn[t]  = sqrtf(tx*tx+ty*ty+tz*tz) + 1e-6f;
        rotn[t] = sqrtf(rx*rx+ry*ry+rz*rz) + 1e-6f;
    }
    __syncthreads();
    for (int idx = t; idx < 2 * BQ * 48; idx += blockDim.x) {
        int head = idx / (BQ * 48);
        int bb = (idx / 48) % BQ;
        int j = idx % 48;
        const float* w1 = head ? rwl1 : twl1;
        const float* b1 = head ? rbl1 : tbl1;
        float nv = head ? rotn[bb] : trn[bb];
        float a = b1[j] + nv * w1[j];                   // k=0 row is the norm input
        for (int k = 0; k < 32; ++k) a += sig[bb * 32 + k] * w1[(1 + k) * 48 + j];
        hid[idx] = fmaxf(a, 0.0f);
    }
    __syncthreads();
    if (t < 2 * BQ) {
        int head = t / BQ, bb = t % BQ;
        const float* w2 = head ? rwl2 : twl2;
        float a = (head ? rbl2 : tbl2)[0];
        for (int j = 0; j < 48; ++j) a += hid[head * BQ * 48 + bb * 48 + j] * w2[j];
        scale[t] = a;
    }
    __syncthreads();
    if (t < BQ * 6) {
        int bb = t / 6, c = t % 6;
        float v = (c < 3) ? tr[bb * 3 + c] / trn[bb] * scale[bb]
                          : rot[bb * 3 + (c - 3)] / rotn[bb] * scale[BQ + bb];
        out[t] = v;
    }
}

// ---------------- host launcher ----------------
// d_in layout assumption: top-level setup_inputs() dict order, nested `params`
// flattened pytree-style (alphabetical keys, 'b' before 'w'):
// 0 lig_pos 1 rec_pos 2 sigma_emb 3 lig_edge_attr
// 4..7 center_edge(l1b,l1w,l2b,l2w) 8..11 conv0 12..15 conv1 16..19 cross_edge
// 20..23 final_conv 24..27 lig_edge 28 lig_emb 29..30 lig_sigma(b,w)
// 31..34 rec_edge 35 rec_emb 36..39 rec_sigma 40..43 rot_head 44..47 tr_head
// 48 lig_idx 49 rec_idx 50 lig_batch 51 rec_batch
// 52 lig_edge_index 53 rec_edge_index 54 cross_edge_index
extern "C" void kernel_launch(void* const* d_in, const int* in_sizes, int n_in,
                              void* d_out, int out_size, void* d_ws, size_t ws_size,
                              hipStream_t stream) {
    (void)in_sizes; (void)n_in; (void)out_size; (void)ws_size;
    const float* lig_pos   = (const float*)d_in[0];
    const float* rec_pos   = (const float*)d_in[1];
    const float* sigma     = (const float*)d_in[2];
    const float* lig_eattr = (const float*)d_in[3];
    const float* ce_l1b = (const float*)d_in[4];  const float* ce_l1w = (const float*)d_in[5];
    const float* ce_l2b = (const float*)d_in[6];  const float* ce_l2w = (const float*)d_in[7];
    const float* c0_l1b = (const float*)d_in[8];  const float* c0_l1w = (const float*)d_in[9];
    const float* c0_l2b = (const float*)d_in[10]; const float* c0_l2w = (const float*)d_in[11];
    const float* c1_l1b = (const float*)d_in[12]; const float* c1_l1w = (const float*)d_in[13];
    const float* c1_l2b = (const float*)d_in[14]; const float* c1_l2w = (const float*)d_in[15];
    const float* cr_l1b = (const float*)d_in[16]; const float* cr_l1w = (const float*)d_in[17];
    const float* cr_l2b = (const float*)d_in[18]; const float* cr_l2w = (const float*)d_in[19];
    const float* fc_l1b = (const float*)d_in[20]; const float* fc_l1w = (const float*)d_in[21];
    const float* fc_l2b = (const float*)d_in[22]; const float* fc_l2w = (const float*)d_in[23];
    const float* le_l1b = (const float*)d_in[24]; const float* le_l1w = (const float*)d_in[25];
    const float* le_l2b = (const float*)d_in[26]; const float* le_l2w = (const float*)d_in[27];
    const float* lig_emb = (const float*)d_in[28];
    const float* ls_b = (const float*)d_in[29];   const float* ls_w = (const float*)d_in[30];
    const float* re_l1b = (const float*)d_in[31]; const float* re_l1w = (const float*)d_in[32];
    const float* re_l2b = (const float*)d_in[33]; const float* re_l2w = (const float*)d_in[34];
    const float* rec_emb = (const float*)d_in[35];
    const float* rs_l1b = (const float*)d_in[36]; const float* rs_l1w = (const float*)d_in[37];
    const float* rs_l2b = (const float*)d_in[38]; const float* rs_l2w = (const float*)d_in[39];
    const float* rh_l1b = (const float*)d_in[40]; const float* rh_l1w = (const float*)d_in[41];
    const float* rh_l2b = (const float*)d_in[42]; const float* rh_l2w = (const float*)d_in[43];
    const float* th_l1b = (const float*)d_in[44]; const float* th_l1w = (const float*)d_in[45];
    const float* th_l2b = (const float*)d_in[46]; const float* th_l2w = (const float*)d_in[47];
    const int* lig_idx   = (const int*)d_in[48];
    const int* rec_idx   = (const int*)d_in[49];
    const int* lig_batch = (const int*)d_in[50];
    const int* rec_batch = (const int*)d_in[51];
    const int* lig_ei    = (const int*)d_in[52];
    const int* rec_ei    = (const int*)d_in[53];
    const int* cross_ei  = (const int*)d_in[54];
    float* out = (float*)d_out;

    // -------- workspace carve-out (~67 MB, fixed offsets, 256B aligned) --------
    char* wp = (char*)d_ws;
    auto alloc = [&](size_t bytes) -> void* {
        void* r = (void*)wp; wp += (bytes + 255) & ~(size_t)255; return r;
    };
    float* x    = (float*)alloc(sizeof(float) * NNV * XD);
    float* agg  = (float*)alloc(sizeof(float) * NNV * XD);
    float* cnt  = (float*)alloc(sizeof(float) * NNV);
    int*   src  = (int*)alloc(sizeof(int) * ETV);
    int*   dst  = (int*)alloc(sizeof(int) * ETV);
    float* darr = (float*)alloc(sizeof(float) * ETV);
    float* sh1  = (float*)alloc(sizeof(float) * ETV * 3);
    float* ea   = (float*)alloc(sizeof(float) * (size_t)ETV * NSV);
    float* ligv = (float*)alloc(sizeof(float) * BQ * NSV);
    float* recv = (float*)alloc(sizeof(float) * BQ * NSV);
    float* in_l = (float*)alloc(sizeof(float) * ELV * 68);
    float* in_r = (float*)alloc(sizeof(float) * ERV * 32);
    float* in_c = (float*)alloc(sizeof(float) * ECV * 64);
    float* tmp  = (float*)alloc(sizeof(float) * ECV * 48);
    unsigned short* ef16 = (unsigned short*)alloc(sizeof(unsigned short) * (size_t)ETV * KP);
    unsigned short* h16  = (unsigned short*)alloc(sizeof(unsigned short) * (size_t)ETV * KP);
    unsigned short* w1a  = (unsigned short*)alloc(sizeof(unsigned short) * KP * NP1);
    unsigned short* w2a  = (unsigned short*)alloc(sizeof(unsigned short) * KP * NW0);
    unsigned short* w1b  = (unsigned short*)alloc(sizeof(unsigned short) * KP * NP1);
    unsigned short* w2b  = (unsigned short*)alloc(sizeof(unsigned short) * KP * NW1P);
    float* b1a = (float*)alloc(sizeof(float) * NP1);
    float* b2a = (float*)alloc(sizeof(float) * NW0);
    float* b1b = (float*)alloc(sizeof(float) * NP1);
    float* b2b = (float*)alloc(sizeof(float) * NW1P);
    float* wchunk = (float*)alloc(sizeof(float) * (size_t)CHUNKE * NW1P); // 31 MB << 192 MB L2
    float* center = (float*)alloc(sizeof(float) * BQ * 3);
    float* ccnt   = (float*)alloc(sizeof(float) * BQ);
    float* cs1    = (float*)alloc(sizeof(float) * NLV * 3);
    float* fin_in = (float*)alloc(sizeof(float) * NLV * 64);
    float* cea_h  = (float*)alloc(sizeof(float) * NLV * 48);
    float* cea    = (float*)alloc(sizeof(float) * NLV * 48);
    float* wf_in  = (float*)alloc(sizeof(float) * NLV * 96);
    float* wf_h   = (float*)alloc(sizeof(float) * NLV * 96);
    float* wf     = (float*)alloc(sizeof(float) * NLV * 176);
    float* gacc   = (float*)alloc(sizeof(float) * BQ * 12);
    float* gcnt   = (float*)alloc(sizeof(float) * BQ);

    auto g1 = [](long n) { return dim3((unsigned)((n + 255) / 256)); };

    // ---- node embeddings ----
    k_zero<<<g1(NNV), 256, 0, stream>>>(cnt, NNV);
    k_batch_vecs<<<1, 256, 0, stream>>>(sigma, ls_w, ls_b, rs_l1w, rs_l1b, rs_l2w, rs_l2b, ligv, recv);
    k_node_init<<<g1((long)NNV * XD), 256, 0, stream>>>(lig_emb, rec_emb, lig_idx, rec_idx,
                                                        lig_batch, rec_batch, ligv, recv, x);
    // ---- edge geometry ----
    k_edges<<<g1(ETV), 256, 0, stream>>>(lig_pos, rec_pos, lig_ei, rec_ei, cross_ei,
                                         src, dst, darr, sh1, cnt);
    // ---- edge feature MLPs -> ea[ETV, 48] ----
    k_in_l<<<g1(ELV), 256, 0, stream>>>(lig_eattr, sigma, lig_ei, lig_batch, darr, in_l);
    k_in_r<<<g1(ERV), 256, 0, stream>>>(darr, in_r);
    k_in_c<<<g1(ECV), 256, 0, stream>>>(sigma, cross_ei, lig_batch, darr, in_c);
    k_linear<<<g1((long)ELV * 48), 256, 0, stream>>>(in_l, le_l1w, le_l1b, tmp, ELV, 68, 48, 1);
    k_linear<<<g1((long)ELV * 48), 256, 0, stream>>>(tmp, le_l2w, le_l2b, ea, ELV, 48, 48, 0);
    k_linear<<<g1((long)ERV * 48), 256, 0, stream>>>(in_r, re_l1w, re_l1b, tmp, ERV, 32, 48, 1);
    k_linear<<<g1((long)ERV * 48), 256, 0, stream>>>(tmp, re_l2w, re_l2b, ea + (size_t)ELV * 48, ERV, 48, 48, 0);
    k_linear<<<g1((long)ECV * 48), 256, 0, stream>>>(in_c, cr_l1w, cr_l1b, tmp, ECV, 64, 48, 1);
    k_linear<<<g1((long)ECV * 48), 256, 0, stream>>>(tmp, cr_l2w, cr_l2b, ea + (size_t)(ELV + ERV) * 48, ECV, 48, 48, 0);
    k_copy<<<g1((long)ECV * 48), 256, 0, stream>>>(ea + (size_t)(ELV + ERV + ECV) * 48,
                                                   ea + (size_t)(ELV + ERV) * 48, ECV * 48);
    // ---- stage conv weights to bf16 fragment-major (zero-padded) ----
    k_stage_w<<<g1((long)KP * NP1), 256, 0, stream>>>(c0_l1w, w1a, 144, 144, NP1);
    k_stage_w<<<g1((long)KP * NW0), 256, 0, stream>>>(c0_l2w, w2a, 144, NW0, NW0);
    k_stage_w<<<g1((long)KP * NP1), 256, 0, stream>>>(c1_l1w, w1b, 144, 144, NP1);
    k_stage_w<<<g1((long)KP * NW1P), 256, 0, stream>>>(c1_l2w, w2b, 144, NW1V, NW1P);
    k_pad_bias<<<1, NP1, 0, stream>>>(c0_l1b, b1a, 144, NP1);
    k_pad_bias<<<g1(NW0), 256, 0, stream>>>(c0_l2b, b2a, NW0, NW0);
    k_pad_bias<<<1, NP1, 0, stream>>>(c1_l1b, b1b, 144, NP1);
    k_pad_bias<<<g1(NW1P), 256, 0, stream>>>(c1_l2b, b2b, NW1V, NW1P);

    // ---- conv0 ----
    k_stage_ef<<<g1((long)ETV * (KP / 8)), 256, 0, stream>>>(ea, x, src, dst, ef16);
    k_wmma_gemm<<<dim3(NP1 / 32, ETV / 64), 256, 0, stream>>>(ef16, KP, w1a, NP1 / 16, b1a,
                                                              nullptr, h16, KP, 1, 1);
    k_zero<<<g1((long)NNV * XD), 256, 0, stream>>>(agg, NNV * XD);
    for (int c = 0; c < NCHUNK; ++c) {
        k_wmma_gemm<<<dim3(NW0 / 32, CHUNKE / 64), 256, 0, stream>>>(
            h16 + (size_t)c * CHUNKE * KP, KP, w2a, NW0 / 16, b2a, wchunk, nullptr, NW0, 0, 0);
        k_tp0<<<CHUNKE, 96, 0, stream>>>(wchunk, NW0, c * CHUNKE, x, src, dst, sh1, agg);
    }
    k_xupdate<<<g1((long)NNV * XD), 256, 0, stream>>>(x, agg, cnt);

    // ---- conv1 ----
    k_stage_ef<<<g1((long)ETV * (KP / 8)), 256, 0, stream>>>(ea, x, src, dst, ef16);
    k_wmma_gemm<<<dim3(NP1 / 32, ETV / 64), 256, 0, stream>>>(ef16, KP, w1b, NP1 / 16, b1b,
                                                              nullptr, h16, KP, 1, 1);
    k_zero<<<g1((long)NNV * XD), 256, 0, stream>>>(agg, NNV * XD);
    for (int c = 0; c < NCHUNK; ++c) {
        k_wmma_gemm<<<dim3(NW1P / 32, CHUNKE / 64), 256, 0, stream>>>(
            h16 + (size_t)c * CHUNKE * KP, KP, w2b, NW1P / 16, b2b, wchunk, nullptr, NW1P, 0, 0);
        k_tp1<<<CHUNKE, 128, 0, stream>>>(wchunk, NW1P, c * CHUNKE, x, src, dst, sh1, agg);
    }
    k_xupdate<<<g1((long)NNV * XD), 256, 0, stream>>>(x, agg, cnt);

    // ---- final pseudotorque conv + heads ----
    k_zero<<<1, 32, 0, stream>>>(center, BQ * 3);
    k_zero<<<1, 32, 0, stream>>>(ccnt, BQ);
    k_center_acc<<<1, 256, 0, stream>>>(lig_pos, lig_batch, center, ccnt);
    k_center_div<<<1, 32, 0, stream>>>(center, ccnt);
    k_final_feats<<<1, 256, 0, stream>>>(lig_pos, lig_batch, center, sigma, cs1, fin_in);
    k_linear<<<g1((long)NLV * 48), 256, 0, stream>>>(fin_in, ce_l1w, ce_l1b, cea_h, NLV, 64, 48, 1);
    k_linear<<<g1((long)NLV * 48), 256, 0, stream>>>(cea_h, ce_l2w, ce_l2b, cea, NLV, 48, 48, 0);
    k_wf_in<<<g1((long)NLV * 96), 256, 0, stream>>>(cea, x, wf_in);
    k_linear<<<g1((long)NLV * 96), 256, 0, stream>>>(wf_in, fc_l1w, fc_l1b, wf_h, NLV, 96, 96, 1);
    k_linear<<<g1((long)NLV * 176), 256, 0, stream>>>(wf_h, fc_l2w, fc_l2b, wf, NLV, 96, 176, 0);
    k_zero<<<1, 64, 0, stream>>>(gacc, BQ * 12);
    k_zero<<<1, 32, 0, stream>>>(gcnt, BQ);
    k_tpf<<<NLV, 12, 0, stream>>>(x, cs1, lig_batch, wf, gacc, gcnt);
    k_heads<<<1, 256, 0, stream>>>(gacc, gcnt, sigma,
                                   th_l1w, th_l1b, th_l2w, th_l2b,
                                   rh_l1w, rh_l1b, rh_l2w, rh_l2b, out);
}
